// CLinear_20272245637517
// MI455X (gfx1250) — matrix operand order repaired and checked by reading, not compiled, not run on hardware
//
#include <hip/hip_runtime.h>
#include <stdint.h>

// out[M,N] = x[M,K] * dequant(qw[N,K])^T + bias
// M = 8192, K = 4096, N = 11008, quant group = 256
typedef _Float16 v16h __attribute__((ext_vector_type(16)));
typedef float    v8f  __attribute__((ext_vector_type(8)));

constexpr int M_DIM   = 4 * 2048;
constexpr int K_DIM   = 4096;
constexpr int N_DIM   = 11008;
constexpr int NGROUPS = 16;        // K / 256

#define TILE_M 128
#define TILE_N 128
#define TILE_K 32
#define LDA    40                  // padded LDS row stride (halves): 80 B = 5x16B chunks
#define BUF_HALVES (TILE_M * LDA)  // halves per LDS buffer

// ---------------- CDNA5 async copy helpers ----------------
// Pass the real LDS pointer into the asm so it ESCAPES (otherwise LLVM proves
// the LDS array is never written and folds the ds_loads to undef).  The VDST
// operand is the LDS byte address: low 32 bits of the generic LDS pointer.
__device__ __forceinline__ void async_copy_b128(_Float16* lds, const _Float16* g) {
    uint32_t loff = (uint32_t)(unsigned long long)lds;   // LDS byte offset
    unsigned long long ga = (unsigned long long)g;
    asm volatile("global_load_async_to_lds_b128 %0, %1, off"
                 :: "v"(loff), "v"(ga), "v"(lds)         // %2 unused: escapes LDS ptr
                 : "memory");
}
__device__ __forceinline__ void wait_async0() {
#if __has_builtin(__builtin_amdgcn_s_wait_asynccnt)
    __builtin_amdgcn_s_wait_asynccnt(0);
#else
    asm volatile("s_wait_asynccnt 0" ::: "memory");
#endif
}

// ---------------- prepass: x fp32 -> f16 ----------------
__global__ __launch_bounds__(256)
void convert_x_kernel(const float* __restrict__ x, _Float16* __restrict__ a16) {
    size_t i = ((size_t)blockIdx.x * 256 + threadIdx.x) * 8;
    float4 f0 = *(const float4*)(x + i);
    float4 f1 = *(const float4*)(x + i + 4);
    union { _Float16 h[8]; uint4 u; } st;
    st.h[0] = (_Float16)f0.x; st.h[1] = (_Float16)f0.y;
    st.h[2] = (_Float16)f0.z; st.h[3] = (_Float16)f0.w;
    st.h[4] = (_Float16)f1.x; st.h[5] = (_Float16)f1.y;
    st.h[6] = (_Float16)f1.z; st.h[7] = (_Float16)f1.w;
    *(uint4*)(a16 + i) = st.u;
}

// ---------------- prepass: int8 weights -> f16 (scale folded) ----------------
__global__ __launch_bounds__(256)
void convert_w_kernel(const int8_t* __restrict__ qw, const float* __restrict__ scale,
                      _Float16* __restrict__ w16) {
    size_t c  = (size_t)blockIdx.x * 256 + threadIdx.x;   // one 16-byte chunk per thread
    int    o  = (int)(c >> 8);                            // K/16 = 256 chunks per row
    int    kc = (int)(c & 255) * 16;
    int    g  = kc >> 8;
    float invs = 1.0f / scale[o * NGROUPS + g];
    union { int4 i4; int8_t b[16]; } q;
    q.i4 = *(const int4*)(qw + (size_t)o * K_DIM + kc);
    union { _Float16 h[16]; uint4 u[2]; } st;
#pragma unroll
    for (int i = 0; i < 16; ++i)
        st.h[i] = (_Float16)((float)q.b[i] * invs);
    *(uint4*)(w16 + (size_t)o * K_DIM + kc)     = st.u[0];
    *(uint4*)(w16 + (size_t)o * K_DIM + kc + 8) = st.u[1];
}

// ---------------- main GEMM: all-f16 operands, async double-buffered A ----------------
__global__ __launch_bounds__(256)
void gemm_f16_wmma_kernel(const _Float16* __restrict__ a16,
                          const _Float16* __restrict__ w16,
                          const float* __restrict__ bias,
                          float* __restrict__ out)
{
    __shared__ _Float16 As[2 * BUF_HALVES];

    const int tid    = threadIdx.x;
    const int lane   = tid & 31;
    const int wid    = tid >> 5;
    const int wave_m = wid >> 2;                // 0..1 -> 64 rows
    const int wave_n = wid & 3;                 // 0..3 -> 32 cols

    const int blockN = blockIdx.x * TILE_N;
    const int blockM = blockIdx.y * TILE_M;

    // A staging: 512 16-byte chunks per tile, 2 per thread (pure async copy)
    const int c0 = tid * 2, c1 = tid * 2 + 1;
    const int r0 = c0 >> 2, h0 = c0 & 3;
    const int r1 = c1 >> 2, h1 = c1 & 3;
    const _Float16* ag0 = a16 + (size_t)(blockM + r0) * K_DIM + h0 * 8;
    const _Float16* ag1 = a16 + (size_t)(blockM + r1) * K_DIM + h1 * 8;
    _Float16* al0 = &As[r0 * LDA + h0 * 8];
    _Float16* al1 = &As[r1 * LDA + h1 * 8];

    // fragment lane coordinates (CDNA5 16-bit WMMA layouts)
    const int nl    = lane & 15;
    const int khalf = (lane >> 4) * 8;          // unused: A-frag also uses kk {0,16}
    const int kk    = (lane >> 4) * 16;         // B-frag k sub-offset {0,16}

    const _Float16* wp0 = w16 + (size_t)(blockN + wave_n * 32 + nl) * K_DIM + kk;
    const _Float16* wp1 = wp0 + (size_t)16 * K_DIM;

    v8f acc[4][2];
#pragma unroll
    for (int i = 0; i < 4; ++i)
#pragma unroll
        for (int j = 0; j < 2; ++j)
#pragma unroll
            for (int e = 0; e < 8; ++e)
                acc[i][j][e] = 0.0f;

    // prologue: async-stage tile 0 into buffer 0; preload B frags for k0=0
    async_copy_b128(al0, ag0);
    async_copy_b128(al1, ag1);

    v16h bfrag[2];
    {
        union { uint4 u[2]; v16h h; } ld;
        ld.u[0] = *(const uint4*)(wp0);     ld.u[1] = *(const uint4*)(wp0 + 8);
        bfrag[0] = ld.h;
        ld.u[0] = *(const uint4*)(wp1);     ld.u[1] = *(const uint4*)(wp1 + 8);
        bfrag[1] = ld.h;
    }

#pragma unroll 2
    for (int k0 = 0; k0 < K_DIM; k0 += TILE_K) {
        const int cur  = (k0 >> 5) & 1;
        const bool more = (k0 + TILE_K) < K_DIM;
        const int kn   = more ? (k0 + TILE_K) : k0;   // clamped prefetch index

        wait_async0();          // this wave's chunks of tile t are in LDS
        __syncthreads();        // everyone's chunks are in LDS

        if (more) {             // async-prefetch A tile t+1 into the other buffer
            async_copy_b128(al0 + (cur ^ 1) * BUF_HALVES, ag0 + kn);
            async_copy_b128(al1 + (cur ^ 1) * BUF_HALVES, ag1 + kn);
        }

        // prefetch next B fragments (overlaps the WMMAs below)
        v16h bnext[2];
        {
            union { uint4 u[2]; v16h h; } ld;
            ld.u[0] = *(const uint4*)(wp0 + kn);  ld.u[1] = *(const uint4*)(wp0 + kn + 8);
            bnext[0] = ld.h;
            ld.u[0] = *(const uint4*)(wp1 + kn);  ld.u[1] = *(const uint4*)(wp1 + kn + 8);
            bnext[1] = ld.h;
        }

        // A fragments from current LDS buffer
        const _Float16* abuf = &As[cur * BUF_HALVES];
        v16h afrag[4];
#pragma unroll
        for (int fm = 0; fm < 4; ++fm) {
            const int row = wave_m * 64 + fm * 16 + nl;
            const _Float16* ap = abuf + row * LDA;
            union { uint4 u[2]; v16h h; } ld;
            ld.u[0] = *(const uint4*)(ap + kk);           // K = kk+0..7
            ld.u[1] = *(const uint4*)(ap + kk + 8);       // K = kk+8..15
            afrag[fm] = ld.h;
        }

#pragma unroll
        for (int fm = 0; fm < 4; ++fm)
#pragma unroll
            for (int fn = 0; fn < 2; ++fn)
                acc[fm][fn] = __builtin_amdgcn_wmma_f32_16x16x32_f16(
                    false, afrag[fm], false, bfrag[fn],
                    (short)0, acc[fm][fn], false, false);

        bfrag[0] = bnext[0];
        bfrag[1] = bnext[1];
    }

    const int mh = (lane >> 4) * 8;             // C/D: VGPR r holds M = mh + r
#pragma unroll
    for (int fm = 0; fm < 4; ++fm) {
#pragma unroll
        for (int fn = 0; fn < 2; ++fn) {
            const int o = blockN + wave_n * 32 + fn * 16 + nl;
            const float bv = bias[o];
#pragma unroll
            for (int r = 0; r < 8; ++r) {
                const int m = blockM + wave_m * 64 + fm * 16 + mh + r;
                out[(size_t)m * N_DIM + o] = acc[fm][fn][r] + bv;
            }
        }
    }
}

// ---------------- fallback: fused kernel (used if ws is too small) ----------------
__global__ __launch_bounds__(256)
void clinear_wmma_fused(const float* __restrict__ x,
                        const int8_t* __restrict__ qw,
                        const float* __restrict__ scale,
                        const float* __restrict__ bias,
                        float* __restrict__ out)
{
    __shared__ _Float16 As[TILE_M * LDA];

    const int tid    = threadIdx.x;
    const int lane   = tid & 31;
    const int wid    = tid >> 5;
    const int wave_m = wid >> 2;
    const int wave_n = wid & 3;
    const int blockN = blockIdx.x * TILE_N;
    const int blockM = blockIdx.y * TILE_M;

    const int ar = tid >> 1;
    const int ac = (tid & 1) * 16;
    const float* xrow = x + (size_t)(blockM + ar) * K_DIM + ac;

    const int nl    = lane & 15;
    const int khalf = (lane >> 4) * 8;
    const int kk    = (lane >> 4) * 16;

    v8f acc[4][2];
#pragma unroll
    for (int i = 0; i < 4; ++i)
#pragma unroll
        for (int j = 0; j < 2; ++j)
#pragma unroll
            for (int e = 0; e < 8; ++e)
                acc[i][j][e] = 0.0f;

    for (int k0 = 0; k0 < K_DIM; k0 += TILE_K) {
        __syncthreads();
        {
            const float4* xp = (const float4*)(xrow + k0);
            float4 f0 = xp[0], f1 = xp[1], f2 = xp[2], f3 = xp[3];
            union { _Float16 h[16]; uint4 u[2]; } st;
            st.h[0]  = (_Float16)f0.x; st.h[1]  = (_Float16)f0.y;
            st.h[2]  = (_Float16)f0.z; st.h[3]  = (_Float16)f0.w;
            st.h[4]  = (_Float16)f1.x; st.h[5]  = (_Float16)f1.y;
            st.h[6]  = (_Float16)f1.z; st.h[7]  = (_Float16)f1.w;
            st.h[8]  = (_Float16)f2.x; st.h[9]  = (_Float16)f2.y;
            st.h[10] = (_Float16)f2.z; st.h[11] = (_Float16)f2.w;
            st.h[12] = (_Float16)f3.x; st.h[13] = (_Float16)f3.y;
            st.h[14] = (_Float16)f3.z; st.h[15] = (_Float16)f3.w;
            *(uint4*)&As[ar * LDA + ac]     = st.u[0];
            *(uint4*)&As[ar * LDA + ac + 8] = st.u[1];
        }
        __syncthreads();

        const int g = k0 >> 8;
        v16h bfrag[2];
#pragma unroll
        for (int fn = 0; fn < 2; ++fn) {
            const int o = blockN + wave_n * 32 + fn * 16 + nl;
            const float invs = 1.0f / scale[o * NGROUPS + g];
            union { int4 i4; int8_t b[16]; } q;
            q.i4 = *(const int4*)(qw + (size_t)o * K_DIM + k0 + kk);
#pragma unroll
            for (int i = 0; i < 16; ++i)
                bfrag[fn][i] = (_Float16)((float)q.b[i] * invs);
        }

        v16h afrag[4];
#pragma unroll
        for (int fm = 0; fm < 4; ++fm) {
            const int row = wave_m * 64 + fm * 16 + nl;
            const _Float16* ap = &As[row * LDA];
            union { uint4 u[2]; v16h h; } ld;
            ld.u[0] = *(const uint4*)(ap + kk);
            ld.u[1] = *(const uint4*)(ap + kk + 8);
            afrag[fm] = ld.h;
        }

#pragma unroll
        for (int fm = 0; fm < 4; ++fm)
#pragma unroll
            for (int fn = 0; fn < 2; ++fn)
                acc[fm][fn] = __builtin_amdgcn_wmma_f32_16x16x32_f16(
                    false, afrag[fm], false, bfrag[fn],
                    (short)0, acc[fm][fn], false, false);
    }

    const int mh = (lane >> 4) * 8;
#pragma unroll
    for (int fm = 0; fm < 4; ++fm)
#pragma unroll
        for (int fn = 0; fn < 2; ++fn) {
            const int o = blockN + wave_n * 32 + fn * 16 + nl;
            const float bv = bias[o];
#pragma unroll
            for (int r = 0; r < 8; ++r) {
                const int m = blockM + wave_m * 64 + fm * 16 + mh + r;
                out[(size_t)m * N_DIM + o] = acc[fm][fn][r] + bv;
            }
        }
}

extern "C" void kernel_launch(void* const* d_in, const int* in_sizes, int n_in,
                              void* d_out, int out_size, void* d_ws, size_t ws_size,
                              hipStream_t stream) {
    const float*  x  = (const float*)d_in[0];
    const int8_t* qw = (const int8_t*)d_in[1];
    const float*  sc = (const float*)d_in[2];
    const float*  bi = (const float*)d_in[3];
    float* out = (float*)d_out;

    const size_t needA = (size_t)M_DIM * K_DIM * sizeof(_Float16);   // 64 MB
    const size_t needW = (size_t)N_DIM * K_DIM * sizeof(_Float16);   // 86 MB
    dim3 grid(N_DIM / TILE_N, M_DIM / TILE_M);                       // 86 x 64

    if (ws_size >= needA + needW) {
        _Float16* a16 = (_Float16*)d_ws;
        _Float16* w16 = (_Float16*)((char*)d_ws + needA);
        convert_x_kernel<<<(int)((size_t)M_DIM * K_DIM / (256 * 8)), 256, 0, stream>>>(x, a16);
        convert_w_kernel<<<(int)((size_t)N_DIM * K_DIM / (256 * 16)), 256, 0, stream>>>(qw, sc, w16);
        gemm_f16_wmma_kernel<<<grid, 256, 0, stream>>>(a16, w16, bi, out);
    } else {
        clinear_wmma_fused<<<grid, 256, 0, stream>>>(x, qw, sc, bi, out);
    }
}
